// ParallelRetNetLayer_60413009985544
// MI455X (gfx1250) — matrix operand order repaired and checked by reading, not compiled, not run on hardware
//
#include <hip/hip_runtime.h>
#include <hip/hip_bf16.h>

// ---------------------------------------------------------------------------
// ParallelRetNet layer for MI455X (gfx1250), wave32 + WMMA bf16.
//
// Factorization: D[i,j] = gamma^(i-j) = gamma^i * gamma^-j  (non-causal), so
//   y[b] = (gamma^i Q[b]) @ (  (gamma^-j K[b])^T @ V[b]  )
// i.e. three projections + two batched [*,1024,1024] GEMMs + GroupNorm.
// All GEMMs use v_wmma_f32_16x16x32_bf16 with fragments loaded directly from
// global memory following the CDNA5 VGPR layouts (CDNA5 ISA §7.12.2).
// ---------------------------------------------------------------------------

typedef __attribute__((ext_vector_type(16))) __bf16 bf16x16;
typedef __attribute__((ext_vector_type(8)))  __bf16 bf16x8;
typedef __attribute__((ext_vector_type(8)))  float  f32x8;

#define BB 16
#define SS 1024
#define DD 1024
#define LOG2_GAMMA (-0.04580368961312515f)   // log2(0.96875) = log2(31/32)
#define GN_EPS 1e-3f

union AFrag {
    bf16x16 v;
    bf16x8  h[2];
};

// ---------------------------------------------------------------------------
// NT GEMM:  C[m,n] = sum_k A[m,k] * Bt[n,k]
//   A : row-major [M,K] bf16, lda elements
//   Bt: row-major [N,K] bf16 (i.e. B column-major), ldb elements
// Workgroup = 4 waves (128 thr), WG tile 128x128, wave tile 64x64
// (4x4 fragments of 16x16, K-step 32).
// Epilogue: optional per-row scale gamma^(dir * (m & (SS-1))), and either
//   row-major store (bf16 or fp32) or transposed store C^T (bf16, vectorized).
// ---------------------------------------------------------------------------
__global__ __launch_bounds__(128) void gemm_nt_wmma(
    const __bf16* __restrict__ A,  long lda, long aBatch,
    const __bf16* __restrict__ Bt, long ldb, long bBatch,
    void* __restrict__ C,          long ldc, long cBatch,
    int K, float scale_dir, int store_trans, int out_f32)
{
    const int lane = threadIdx.x & 31;
    const int wave = threadIdx.x >> 5;
    const int g    = lane >> 4;      // half-wave group (0/1)
    const int lm   = lane & 15;

    const long mBase = (long)blockIdx.y * 128 + (long)(wave >> 1) * 64;
    const long nBase = (long)blockIdx.x * 128 + (long)(wave & 1)  * 64;

    const __bf16* Ab = A  + (long)blockIdx.z * aBatch;
    const __bf16* Bb = Bt + (long)blockIdx.z * bBatch;

    f32x8 acc[4][4] = {};

    for (int kb = 0; kb < K; kb += 32) {
        AFrag a[4], b[4];
        // A fragment (16x32 bf16): lane holds row m=lm; halves 0-7 at
        // k = kb + g*8 .. +7, halves 8-15 at k = kb + 16 + g*8 .. +7
        #pragma unroll
        for (int mt = 0; mt < 4; ++mt) {
            const __bf16* p = Ab + (mBase + mt * 16 + lm) * lda + kb + g * 8;
            a[mt].h[0] = *(const bf16x8*)(p);
            a[mt].h[1] = *(const bf16x8*)(p + 16);
        }
        // B fragment (32x16 bf16): lane holds column n=lm; halves 0-7 at
        // k = kb + g*8 .. +7, halves 8-15 at k = kb + 16 + g*8 .. +7
        #pragma unroll
        for (int nt = 0; nt < 4; ++nt) {
            const __bf16* p = Bb + (nBase + nt * 16 + lm) * ldb + kb + g * 8;
            b[nt].h[0] = *(const bf16x8*)(p);
            b[nt].h[1] = *(const bf16x8*)(p + 16);
        }
        #pragma unroll
        for (int mt = 0; mt < 4; ++mt)
            #pragma unroll
            for (int nt = 0; nt < 4; ++nt)
                acc[mt][nt] = __builtin_amdgcn_wmma_f32_16x16x32_bf16(
                    false, a[mt].v, false, b[nt].v,
                    (short)0, acc[mt][nt], false, false);
    }

    // Epilogue. C/D layout: VGPR r, lane l -> M = r + 8*(l/16), N = l%16.
    #pragma unroll
    for (int mt = 0; mt < 4; ++mt) {
        #pragma unroll
        for (int nt = 0; nt < 4; ++nt) {
            const long n  = nBase + nt * 16 + lm;
            const long m0 = mBase + mt * 16 + g * 8;
            float vals[8];
            #pragma unroll
            for (int r = 0; r < 8; ++r) {
                float v = acc[mt][nt][r];
                if (scale_dir != 0.0f) {
                    const int i = (int)((m0 + r) & (long)(SS - 1));
                    v *= __builtin_exp2f(scale_dir * (float)i * LOG2_GAMMA);
                }
                vals[r] = v;
            }
            if (store_trans) {
                // C^T: lane holds 8 consecutive M for fixed N -> one b128 store
                bf16x8 pk;
                #pragma unroll
                for (int r = 0; r < 8; ++r) pk[r] = (__bf16)vals[r];
                *(bf16x8*)((__bf16*)C + (long)blockIdx.z * cBatch + n * ldc + m0) = pk;
            } else if (out_f32) {
                float* Cp = (float*)C + (long)blockIdx.z * cBatch;
                #pragma unroll
                for (int r = 0; r < 8; ++r) Cp[(m0 + r) * ldc + n] = vals[r];
            } else {
                __bf16* Cp = (__bf16*)C + (long)blockIdx.z * cBatch;
                #pragma unroll
                for (int r = 0; r < 8; ++r) Cp[(m0 + r) * ldc + n] = (__bf16)vals[r];
            }
        }
    }
}

// fp32 -> bf16 elementwise
__global__ __launch_bounds__(256) void convert_bf16(
    const float* __restrict__ in, __bf16* __restrict__ out, long n)
{
    long i = (long)blockIdx.x * 256 + threadIdx.x;
    if (i < n) out[i] = (__bf16)in[i];
}

// fp32 [d,d] row-major -> bf16 transposed [n,k]
__global__ __launch_bounds__(256) void transpose_bf16(
    const float* __restrict__ in, __bf16* __restrict__ out, int d)
{
    long i = (long)blockIdx.x * 256 + threadIdx.x;
    if (i < (long)d * d) {
        const long k = i / d, n = i % d;
        out[n * (long)d + k] = (__bf16)in[i];
    }
}

// GroupNorm (groups == channels): normalize each channel over the seq axis.
// Block handles 256 consecutive channels of one batch; coalesced column walk.
__global__ __launch_bounds__(256) void groupnorm_seq(
    float* __restrict__ y, const float* __restrict__ sc,
    const float* __restrict__ bt)
{
    const int  d    = blockIdx.x * 256 + threadIdx.x;
    const long base = (long)blockIdx.y * SS * DD + d;
    float sum = 0.f, sq = 0.f;
    for (int i = 0; i < SS; ++i) {
        const float v = y[base + (long)i * DD];
        sum += v;
        sq  += v * v;
    }
    const float mean = sum * (1.0f / SS);
    const float var  = sq * (1.0f / SS) - mean * mean;
    const float inv  = rsqrtf(var + GN_EPS);
    const float s = sc[d], b = bt[d];
    for (int i = 0; i < SS; ++i) {
        const long o = base + (long)i * DD;
        y[o] = (y[o] - mean) * inv * s + b;
    }
}

extern "C" void kernel_launch(void* const* d_in, const int* in_sizes, int n_in,
                              void* d_out, int out_size, void* d_ws, size_t ws_size,
                              hipStream_t stream) {
    const float* x   = (const float*)d_in[0];
    const float* Wq  = (const float*)d_in[1];
    const float* Wk  = (const float*)d_in[2];
    const float* Wv  = (const float*)d_in[3];
    const float* gsc = (const float*)d_in[4];
    const float* gbt = (const float*)d_in[5];
    float* out = (float*)d_out;

    const long BS = (long)BB * SS;

    // Workspace carve-out (~166 MB total)
    char* w = (char*)d_ws;
    __bf16* xb  = (__bf16*)w; w += BS * DD * 2;          // 32 MB
    __bf16* Wqt = (__bf16*)w; w += (long)DD * DD * 2;    //  2 MB
    __bf16* Wkt = (__bf16*)w; w += (long)DD * DD * 2;
    __bf16* Wvt = (__bf16*)w; w += (long)DD * DD * 2;
    __bf16* Qs  = (__bf16*)w; w += BS * DD * 2;          // 32 MB (row-major [BS,D])
    __bf16* Kst = (__bf16*)w; w += BS * DD * 2;          // 32 MB (transposed [D,BS])
    __bf16* Vt  = (__bf16*)w; w += BS * DD * 2;          // 32 MB (transposed [D,BS])
    __bf16* Mt  = (__bf16*)w; w += (long)BB * DD * DD * 2; // 32 MB ([b][D,D] = M^T row-major)

    // 1) precision conversion + weight transposition
    convert_bf16<<<dim3((unsigned)(BS * DD / 256)), dim3(256), 0, stream>>>(x, xb, BS * DD);
    transpose_bf16<<<dim3(DD * DD / 256), dim3(256), 0, stream>>>(Wq, Wqt, DD);
    transpose_bf16<<<dim3(DD * DD / 256), dim3(256), 0, stream>>>(Wk, Wkt, DD);
    transpose_bf16<<<dim3(DD * DD / 256), dim3(256), 0, stream>>>(Wv, Wvt, DD);

    const dim3 blk(128);
    // 2) Qs = gamma^{+i} * (x @ Wq), row-major bf16
    gemm_nt_wmma<<<dim3(DD / 128, (unsigned)(BS / 128), 1), blk, 0, stream>>>(
        xb, DD, 0, Wqt, DD, 0, Qs, DD, 0, DD, +1.0f, 0, 0);
    // 3) Kst = (gamma^{-j} * (x @ Wk))^T, bf16 [D, BS]
    gemm_nt_wmma<<<dim3(DD / 128, (unsigned)(BS / 128), 1), blk, 0, stream>>>(
        xb, DD, 0, Wkt, DD, 0, Kst, BS, 0, DD, -1.0f, 1, 0);
    // 4) Vt = (x @ Wv)^T, bf16 [D, BS]
    gemm_nt_wmma<<<dim3(DD / 128, (unsigned)(BS / 128), 1), blk, 0, stream>>>(
        xb, DD, 0, Wvt, DD, 0, Vt, BS, 0, DD, 0.0f, 1, 0);
    // 5) Mt[b] = V[b]^T @ Ks[b]  (A = Vt_b [D,S], Bt = Kst_b [D,S], K = S)
    //    stored row-major -> Mt[b][d,e] = M[b][e,d], which is exactly the
    //    column-major B operand needed by step 6.
    gemm_nt_wmma<<<dim3(DD / 128, DD / 128, BB), blk, 0, stream>>>(
        Vt, BS, SS, Kst, BS, SS, Mt, DD, (long)DD * DD, SS, 0.0f, 0, 0);
    // 6) y[b] = Qs[b] @ M[b]  -> fp32 into d_out
    gemm_nt_wmma<<<dim3(DD / 128, SS / 128, BB), blk, 0, stream>>>(
        Qs, DD, (long)SS * DD, Mt, DD, (long)DD * DD, out, DD, (long)SS * DD,
        DD, 0.0f, 0, 1);
    // 7) GroupNorm over sequence axis, in place on d_out
    groupnorm_seq<<<dim3(DD / 256, BB), dim3(256), 0, stream>>>(out, gsc, gbt);
}
